// AGRNN_49787260895381
// MI455X (gfx1250) — matrix operand order repaired and checked by reading, not compiled, not run on hardware
//
#include <hip/hip_runtime.h>
#include <hip/hip_bf16.h>

typedef __bf16 bf16_t;
typedef __attribute__((ext_vector_type(2)))  bf16_t v2bf;
typedef __attribute__((ext_vector_type(8)))  bf16_t v8bf;
typedef __attribute__((ext_vector_type(16))) bf16_t v16bf;
typedef __attribute__((ext_vector_type(8)))  float  v8f;
typedef __attribute__((ext_vector_type(2)))  float  v2f;   // trivial float2

#define AS1 __attribute__((address_space(1)))
typedef const AS1 float* gcfp;                 // explicit global pointers so the
typedef const AS1 v2f*   gcf2p;                // gather select stays global_load

#define F_VIS   512
#define F_LANG  300
#define F_SPAT  16
#define F_CAT   1640            // 512+300+16+300+512
#define KSTEP   32
#define NSTEPS  52              // K padded to 1664
#define HIDDEN  256
#define NCLS    13
#define M_BLK   128             // edges per block (8 M-tiles of 16)

#define S1 (F_VIS)                       // 512
#define S2 (F_VIS + F_LANG)              // 812
#define S3 (F_VIS + F_LANG + F_SPAT)     // 828
#define S4 (F_VIS + 2*F_LANG + F_SPAT)   // 1128

// Pre-packed W1 in workspace: [ks][ntile][laneb][16] bf16, WMMA-B lane layout.
#define WSP_ELEMS (NSTEPS * 16 * 32 * 16)     // 425984 bf16 = 832 KB

// LDS: phase 1: USE_WS -> A double-buffer [0,16KB); else A [0,8KB) B [16KB,32KB)
//      phase 2: h-chunk [0,32KB) | W2^T f32 [32KB, 45.3KB)
#define LDS_BYTES (32768 + 13312)

__global__ __launch_bounds__(256)
void pack_W1_kernel(const float* __restrict__ W1, bf16_t* __restrict__ W1p) {
  const int idx = blockIdx.x * 256 + threadIdx.x;
  if (idx >= WSP_ELEMS) return;
  const int elem  = idx & 15;
  const int laneb = (idx >> 4) & 31;
  const int nt    = (idx >> 9) & 15;
  const int ks    = idx >> 13;
  const int kk = (laneb & 16) | elem;          // K within 32-step
  const int n  = nt * 16 + (laneb & 15);
  const int kg = ks * KSTEP + kk;
  const float w = (kg < F_CAT) ? W1[kg * HIDDEN + n] : 0.0f;
  W1p[idx] = (bf16_t)w;
}

template<bool USE_WS>
__global__ __launch_bounds__(256)
void edge_mlp_kernel(const float* __restrict__ node_feat,
                     const float* __restrict__ node_lang,
                     const float* __restrict__ edge_sp,
                     const float* __restrict__ W1,
                     const float* __restrict__ b1,
                     const float* __restrict__ W2,
                     const float* __restrict__ b2,
                     const int*   __restrict__ src_idx,
                     const int*   __restrict__ dst_idx,
                     const bf16_t* __restrict__ W1p,
                     float*       __restrict__ out,
                     int E)
{
  __shared__ __align__(32) unsigned char smem[LDS_BYTES];
  bf16_t* A_lds = (bf16_t*)smem;               // buffers of 4096 bf16 (8KB) each
  bf16_t* B_lds = (bf16_t*)(smem + 16384);     // fallback only: [16][32][16]
  bf16_t* h_lds = (bf16_t*)smem;               // phase 2: [64][256] bf16
  float*  W2t   = (float*)(smem + 32768);      // phase 2: [13][256] f32

  const int  tid  = threadIdx.x;
  const int  lane = tid & 31;
  const int  wav  = tid >> 5;                  // 0..7
  const long e0   = (long)blockIdx.x * M_BLK;

  const gcfp nfG = (gcfp)node_feat;
  const gcfp nlG = (gcfp)node_lang;
  const gcfp esG = (gcfp)edge_sp;

  // A-gather lane split: kl = k-pair (covers 32 K as v2f), eh = edge half.
  const int kl = lane & 15;
  const int eh = lane >> 4;

  // 8 fixed edges per thread for the whole K loop; indices cached in VGPRs.
  int dstA[8], srcA[8], eA[8];
  #pragma unroll
  for (int j = 0; j < 8; ++j) {
    long e = e0 + wav * 16 + eh * 8 + j;
    if (e > (long)E - 1) e = (long)E - 1;      // clamp (keeps EXEC full)
    eA[j] = (int)e; dstA[j] = dst_idx[e]; srcA[j] = src_idx[e];
  }

  // Stage 128 edges x 32 K into Abuf; segment select hoisted (uniform over j),
  // v2f loads never straddle a segment (all boundaries even), u32 offsets.
  auto stageA = [&](int ks, bf16_t* Abuf) {
    const int k = ks * KSTEP + 2 * kl;
    const gcfp tab = (k < S1) ? nfG
                    : (k < S2) ? nlG
                    : (k < S3) ? esG
                    : (k < S4) ? nlG
                    :            nfG;
    const int col = (k < S1) ? k
                   : (k < S2) ? (k - S1)
                   : (k < S3) ? (k - S2)
                   : (k < S4) ? (k - S3)
                   :            (k - S4);
    const int stride = (k < S1) ? F_VIS
                      : (k < S2) ? F_LANG
                      : (k < S3) ? F_SPAT
                      : (k < S4) ? F_LANG
                      :            F_VIS;
    const int who = (k < S2) ? 0 : (k < S3) ? 2 : 1;    // 0=dst 2=edge 1=src
    bf16_t* dst0 = &Abuf[(wav * 16 + eh * 8) * KSTEP + 2 * kl];

    if (__builtin_expect(ks != NSTEPS - 1, 1)) {        // no K padding here
      #pragma unroll
      for (int j = 0; j < 8; ++j) {
        const int row = (who == 2) ? eA[j] : (who == 1) ? srcA[j] : dstA[j];
        const unsigned off = (unsigned)row * (unsigned)stride + (unsigned)col;
        const v2f v = *(gcf2p)(tab + off);
        v2bf pk; pk.x = (bf16_t)v.x; pk.y = (bf16_t)v.y;
        *(v2bf*)(dst0 + j * KSTEP) = pk;
      }
    } else {                                            // last step: pad K>=1640
      const float padmul = (k < F_CAT) ? 1.0f : 0.0f;
      const int   colz   = (k < F_CAT) ? col : 0;
      const int   strz   = (k < F_CAT) ? stride : 0;    // pad -> read tab[0]
      #pragma unroll
      for (int j = 0; j < 8; ++j) {
        const int row = (who == 2) ? eA[j] : (who == 1) ? srcA[j] : dstA[j];
        const unsigned off = (unsigned)row * (unsigned)strz + (unsigned)colz;
        const v2f v = *(gcf2p)(tab + off);
        v2bf pk; pk.x = (bf16_t)(v.x * padmul); pk.y = (bf16_t)(v.y * padmul);
        *(v2bf*)(dst0 + j * KSTEP) = pk;
      }
    }
  };

  // Fallback-only: stage + swizzle W1 tile into LDS.
  auto stageB = [&](int ks) {
    #pragma unroll 8
    for (int i = 0; i < 32; ++i) {
      const int flat = i * 256 + tid;
      const int kk = flat >> 8, n = flat & 255;
      const int kg = ks * KSTEP + kk;
      const float w = (kg < F_CAT) ? W1[(long)kg * HIDDEN + n] : 0.0f;
      B_lds[(n >> 4) * 512 + ((n & 15) | (kk & 16)) * 16 + (kk & 15)] = (bf16_t)w;
    }
  };

  auto loadB = [&](int ks, int q) -> v16bf {
    if constexpr (USE_WS) {
      const unsigned idx = ((unsigned)(ks * 16 + wav * 2 + q) * 32 + (unsigned)lane) * 16u;
      return *(const v16bf*)&W1p[idx];
    } else {
      return *(const v16bf*)&B_lds[(wav * 2 + q) * 512 + lane * 16];
    }
  };

  v8f acc[8][2];
  #pragma unroll
  for (int mt = 0; mt < 8; ++mt) {
    acc[mt][0] = (v8f){0.f,0.f,0.f,0.f,0.f,0.f,0.f,0.f};
    acc[mt][1] = (v8f){0.f,0.f,0.f,0.f,0.f,0.f,0.f,0.f};
  }

  auto compute = [&](const bf16_t* Abuf, v16bf bf0, v16bf bf1) {
    #pragma unroll
    for (int mt = 0; mt < 8; ++mt) {
      // lanes 0-15: K[0..15]; lanes 16-31: K[16..31]
      const bf16_t* ar = &Abuf[(mt * 16 + (lane & 15)) * KSTEP + ((lane >> 4) << 4)];
      v8bf lo = *(const v8bf*)ar;
      v8bf hi = *(const v8bf*)(ar + 8);
      v16bf a = __builtin_shufflevector(lo, hi,
                  0,1,2,3,4,5,6,7,8,9,10,11,12,13,14,15);
      acc[mt][0] = __builtin_amdgcn_wmma_f32_16x16x32_bf16(
          false, a, false, bf0, (short)0, acc[mt][0], false, false);
      acc[mt][1] = __builtin_amdgcn_wmma_f32_16x16x32_bf16(
          false, a, false, bf1, (short)0, acc[mt][1], false, false);
    }
  };

  if constexpr (USE_WS) {
    // Software pipeline: stage ks+1 (A->LDS buf^1, B->regs) while computing ks.
    stageA(0, A_lds);
    v16bf bn0 = loadB(0, 0), bn1 = loadB(0, 1);
    for (int ks = 0; ks < NSTEPS; ++ks) {
      __syncthreads();                         // stage(ks) visible; buf^1 free
      bf16_t* Acur = A_lds + (ks & 1) * 4096;
      const v16bf bc0 = bn0, bc1 = bn1;
      if (ks + 1 < NSTEPS) {
        stageA(ks + 1, A_lds + ((ks + 1) & 1) * 4096);
        bn0 = loadB(ks + 1, 0);
        bn1 = loadB(ks + 1, 1);
      }
      compute(Acur, bc0, bc1);
    }
  } else {
    for (int ks = 0; ks < NSTEPS; ++ks) {
      __syncthreads();
      stageA(ks, A_lds);
      stageB(ks);
      __syncthreads();
      compute(A_lds, loadB(ks, 0), loadB(ks, 1));
    }
  }

  // ---- epilogue in two 64-edge chunks (reuses staging LDS region) ----
  const int mrow = (lane >> 4) << 3;           // 0 or 8
  const int ncol = lane & 15;
  const float bias0 = b1[wav * 32 + ncol];
  const float bias1 = b1[wav * 32 + 16 + ncol];

  for (int half = 0; half < 2; ++half) {
    __syncthreads();                           // region free (WAR)
    #pragma unroll
    for (int q = 0; q < 2; ++q) {
      const int col = wav * 32 + q * 16 + ncol;
      const float bias = q ? bias1 : bias0;
      #pragma unroll
      for (int mtl = 0; mtl < 4; ++mtl) {
        #pragma unroll
        for (int r = 0; r < 8; ++r) {
          float h = acc[half * 4 + mtl][q][r] + bias;
          h = h > 0.0f ? h : 0.0f;
          h_lds[(mtl * 16 + mrow + r) * HIDDEN + col] = (bf16_t)h;
        }
      }
    }
    if (half == 0) {                           // stage W2^T once: [13][256]
      for (int o = tid; o < NCLS * HIDDEN; o += 256)
        W2t[o] = W2[(o & 255) * NCLS + (o >> 8)];
    }
    __syncthreads();

    // second GEMM: [64,256] @ [256,13] + b2, f32 VALU, contiguous dots
    for (int o = tid; o < 64 * NCLS; o += 256) {
      const int el = o / NCLS, c = o % NCLS;
      const bf16_t* hr = &h_lds[el * HIDDEN];
      const float*  wr = &W2t[c * HIDDEN];
      float s = b2[c];
      #pragma unroll 8
      for (int kk = 0; kk < HIDDEN; ++kk)
        s += (float)hr[kk] * wr[kk];
      const long e = e0 + half * 64 + el;
      if (e < (long)E) out[e * NCLS + c] = s;
    }
  }
}

extern "C" void kernel_launch(void* const* d_in, const int* in_sizes, int n_in,
                              void* d_out, int out_size, void* d_ws, size_t ws_size,
                              hipStream_t stream) {
  (void)n_in; (void)out_size;
  const float* node_feat = (const float*)d_in[0];
  const float* node_lang = (const float*)d_in[1];
  const float* edge_sp   = (const float*)d_in[2];
  const float* W1        = (const float*)d_in[3];
  const float* b1        = (const float*)d_in[4];
  const float* W2        = (const float*)d_in[5];
  const float* b2        = (const float*)d_in[6];
  const int*   src_idx   = (const int*)d_in[7];
  const int*   dst_idx   = (const int*)d_in[8];
  float* out = (float*)d_out;

  const int E = in_sizes[7];
  const int blocks = (E + M_BLK - 1) / M_BLK;  // 1563 for E=200000
  const size_t need = (size_t)WSP_ELEMS * sizeof(bf16_t);

  if (d_ws != nullptr && ws_size >= need) {
    bf16_t* W1p = (bf16_t*)d_ws;
    pack_W1_kernel<<<(WSP_ELEMS + 255) / 256, 256, 0, stream>>>(W1, W1p);
    edge_mlp_kernel<true><<<blocks, 256, 0, stream>>>(
        node_feat, node_lang, edge_sp, W1, b1, W2, b2, src_idx, dst_idx,
        W1p, out, E);
  } else {
    edge_mlp_kernel<false><<<blocks, 256, 0, stream>>>(
        node_feat, node_lang, edge_sp, W1, b1, W2, b2, src_idx, dst_idx,
        nullptr, out, E);
  }
}
